// GNNModel_85555748536481
// MI455X (gfx1250) — compile-verified, matrix operand order checked by reading, not run on hardware
//
#include <hip/hip_runtime.h>
#include <math.h>

// ---------------------------------------------------------------------------
// GATv2 GNN (N=50000, E=800000, G=64) for MI455X / gfx1250.
// GEMMs via v_wmma_f32_16x16x32_bf16: weights pre-packed into fragment order
// (contiguous b128 loads per lane), 4 N-tiles per wave for A reuse + WMMA ILP.
// Edge phase: edge-parallel float4 gathers + float atomics (node features are
// L2-resident: 51MB << 192MB L2).
// ---------------------------------------------------------------------------

#define NN    50000
#define NE    800000
#define ETOT  (NE + NN)          // edges + self loops
#define GG    64
#define INDIM 128
#define HID   64
#define HEADS 4
#define D1    (HID * HEADS)      // 256
#define EPSF  1e-5f
#define SLOPE 0.2f

typedef __bf16 bf16;
typedef __attribute__((ext_vector_type(16))) __bf16 v16bf;
typedef __attribute__((ext_vector_type(8)))  float  v8f;

__device__ __forceinline__ bf16 f2bf(float f) {
  unsigned u = __float_as_uint(f);
  u += 0x7FFFu + ((u >> 16) & 1u);        // round-to-nearest-even
  unsigned short s = (unsigned short)(u >> 16);
  bf16 r;
  __builtin_memcpy(&r, &s, 2);
  return r;
}

__device__ __forceinline__ void atomicMaxF(float* addr, float val) {
  int old = __float_as_int(*addr);
  while (__int_as_float(old) < val) {
    int prev = atomicCAS((int*)addr, old, __float_as_int(val));
    if (prev == old) break;
    old = prev;
  }
}

__device__ __forceinline__ float eluf(float v) {
  return v > 0.f ? v : (__expf(v) - 1.f);
}

// ------------------------------- utility -----------------------------------

__global__ void fill_f32(float* __restrict__ p, float v, int n) {
  int i = blockIdx.x * blockDim.x + threadIdx.x;
  if (i < n) p[i] = v;
}

__global__ void cvt_f32_bf16(const float* __restrict__ in, bf16* __restrict__ out, int n) {
  int i = blockIdx.x * blockDim.x + threadIdx.x;
  if (i < n) out[i] = f2bf(in[i]);
}

// -------------------- weight pack: W[K,Nout] -> fragment order --------------
// Packed layout: [kc][tn][lane][16], so each lane's 16-element B fragment for
// (k-chunk kc, n-tile tn) is 32 contiguous bytes.  Fragment element i holds
// B[kk, n] with kk = kc*32 + (i>>3)*16 + hi + (i&7), n = tn*16 + (lane&15),
// hi = (lane>=16) ? 8 : 0  (per ISA 16-bit 32x16 B layout, mirroring A).
__global__ void pack_b_frag(const float* __restrict__ W, bf16* __restrict__ Wp,
                            int K, int Nout) {
  int idx = blockIdx.x * blockDim.x + threadIdx.x;
  int total = (K >> 5) * (Nout >> 4) * 512;
  if (idx >= total) return;
  int i    = idx & 15;
  int lane = (idx >> 4) & 31;
  int t    = idx >> 9;
  int tiles_n = Nout >> 4;
  int tn = t % tiles_n;
  int kc = t / tiles_n;
  int hi = (lane & 16) ? 8 : 0;
  int kk = (kc << 5) + ((i >> 3) << 4) + hi + (i & 7);
  int n  = (tn << 4) + (lane & 15);
  Wp[idx] = f2bf(W[(size_t)kk * Nout + n]);
}

// ------------------------- WMMA bf16 GEMM + bias ---------------------------
// C[M,Nout] = A[M,K] @ W[K,Nout] + bias.  One 16x64 strip (4 n-tiles) per
// wave32: one A fragment feeds 4 independent WMMAs per k-step.
__global__ __launch_bounds__(128) void wmma_gemm_bias4(
    const bf16* __restrict__ A, const bf16* __restrict__ Wp,
    const float* __restrict__ bias, float* __restrict__ C,
    int M, int K, int Nout)
{
  int wave = blockIdx.x * (blockDim.x >> 5) + (threadIdx.x >> 5);
  int lane = threadIdx.x & 31;
  int tiles_n = Nout >> 4;
  int ngroups = tiles_n >> 2;                   // groups of 4 n-tiles
  int tm = wave / ngroups;
  int tg = wave - tm * ngroups;
  if (tm * 16 >= M) return;                     // wave-uniform exit (EXEC all-1 for WMMA)
  int tn0 = tg << 2;

  int m  = tm * 16 + (lane & 15);
  int hi = (lane & 16) ? 8 : 0;

  v8f acc[4] = {};
  int nkc = K >> 5;
  for (int kc = 0; kc < nkc; ++kc) {
    int k0 = kc << 5;
    v16bf a;
#pragma unroll
    for (int i = 0; i < 16; ++i)                // two contiguous 8-elem runs -> 2x b128
      a[i] = A[(size_t)m * K + k0 + ((i >> 3) << 4) + hi + (i & 7)];
#pragma unroll
    for (int t = 0; t < 4; ++t) {
      const v16bf* pb = (const v16bf*)(Wp +
          ((((size_t)kc * tiles_n) + tn0 + t) * 32 + lane) * 16);
      v16bf b = *pb;                            // 32 contiguous bytes
      acc[t] = __builtin_amdgcn_wmma_f32_16x16x32_bf16(
          false, a, false, b, (short)0, acc[t], false, false);
    }
  }

  int rowbase = tm * 16 + ((lane & 16) ? 8 : 0);
#pragma unroll
  for (int t = 0; t < 4; ++t) {
    int n = ((tn0 + t) << 4) + (lane & 15);
    float bb = bias[n];
#pragma unroll
    for (int r = 0; r < 8; ++r)
      C[(size_t)(rowbase + r) * Nout + n] = acc[t][r] + bb;
  }
}

// ------------------------------ edge phase ---------------------------------

__device__ __forceinline__ void edge_ends(int e, const int* __restrict__ ei,
                                          int& src, int& dst) {
  if (e < NE) { src = ei[e]; dst = ei[NE + e]; }
  else        { src = dst = e - NE; }
}

// logits[e,h] = sum_c att[h,c] * leaky_relu(xl[src,h,c] + xr[dst,h,c]); atomic max into m[dst,h]
__global__ void edge_logits(const float* __restrict__ xl, const float* __restrict__ xr,
                            const float* __restrict__ att, const int* __restrict__ ei,
                            float* __restrict__ logits, float* __restrict__ mmax, int heads) {
  int idx = blockIdx.x * blockDim.x + threadIdx.x;
  if (idx >= ETOT * heads) return;
  int e = idx / heads, h = idx - e * heads;
  int src, dst; edge_ends(e, ei, src, dst);
  int hc = heads * HID;
  const float4* pl = (const float4*)(xl + (size_t)src * hc + h * HID);
  const float4* pr = (const float4*)(xr + (size_t)dst * hc + h * HID);
  const float4* pa = (const float4*)(att + h * HID);
  float s = 0.f;
#pragma unroll 4
  for (int q = 0; q < HID / 4; ++q) {
    float4 l = pl[q], r = pr[q], a = pa[q];
    float v;
    v = l.x + r.x; v = v > 0.f ? v : SLOPE * v; s += a.x * v;
    v = l.y + r.y; v = v > 0.f ? v : SLOPE * v; s += a.y * v;
    v = l.z + r.z; v = v > 0.f ? v : SLOPE * v; s += a.z * v;
    v = l.w + r.w; v = v > 0.f ? v : SLOPE * v; s += a.w * v;
  }
  logits[idx] = s;
  atomicMaxF(mmax + (size_t)dst * heads + h, s);
}

// a = exp(logit - m[dst]); denom[dst,h] += a
__global__ void edge_exp(const float* __restrict__ logits, const float* __restrict__ mmax,
                         const int* __restrict__ ei, float* __restrict__ aout,
                         float* __restrict__ denom, int heads) {
  int idx = blockIdx.x * blockDim.x + threadIdx.x;
  if (idx >= ETOT * heads) return;
  int e = idx / heads, h = idx - e * heads;
  int src, dst; edge_ends(e, ei, src, dst);
  float v = __expf(logits[idx] - mmax[(size_t)dst * heads + h]);
  aout[idx] = v;
  atomicAdd(denom + (size_t)dst * heads + h, v);
}

// out[dst, :] += (a/denom[dst]) * xl[src, :]   (float4 gather, 4 scalar atomics)
__global__ void edge_scatter(const float* __restrict__ aexp, const float* __restrict__ denom,
                             const float* __restrict__ xl, const int* __restrict__ ei,
                             float* __restrict__ out, int heads) {
  int hc = heads * HID, qn = hc >> 2;
  int idx = blockIdx.x * blockDim.x + threadIdx.x;
  if (idx >= ETOT * qn) return;
  int e = idx / qn, q = idx - e * qn;
  int c0 = q << 2, h = c0 / HID;
  int src, dst; edge_ends(e, ei, src, dst);
  float alpha = aexp[(size_t)e * heads + h] / denom[(size_t)dst * heads + h];
  float4 v = *(const float4*)(xl + (size_t)src * hc + c0);
  float* o = out + (size_t)dst * hc + c0;
  atomicAdd(o + 0, alpha * v.x);
  atomicAdd(o + 1, alpha * v.y);
  atomicAdd(o + 2, alpha * v.z);
  atomicAdd(o + 3, alpha * v.w);
}

// ------------------------------- batchnorm ---------------------------------

__global__ __launch_bounds__(256) void bn_stats(const float* __restrict__ x, int n, int cdim,
                                                float* __restrict__ mu, float* __restrict__ rinv) {
  int c = blockIdx.x, tid = threadIdx.x;
  float s = 0.f, s2 = 0.f;
  for (int i = tid; i < n; i += 256) {
    float v = x[(size_t)i * cdim + c];
    s += v; s2 += v * v;
  }
  __shared__ float sh[256], sh2[256];
  sh[tid] = s; sh2[tid] = s2; __syncthreads();
  for (int o = 128; o > 0; o >>= 1) {
    if (tid < o) { sh[tid] += sh[tid + o]; sh2[tid] += sh2[tid + o]; }
    __syncthreads();
  }
  if (tid == 0) {
    float m = sh[0] / n, var = sh2[0] / n - m * m;
    mu[c] = m; rinv[c] = rsqrtf(var + EPSF);
  }
}

__global__ void bn_elu_bf16(const float* __restrict__ x, const float* __restrict__ mu,
                            const float* __restrict__ rinv, const float* __restrict__ gamma,
                            const float* __restrict__ beta, bf16* __restrict__ out,
                            int n, int cdim) {
  int idx = blockIdx.x * blockDim.x + threadIdx.x;
  if (idx >= n * cdim) return;
  int c = idx % cdim;
  out[idx] = f2bf(eluf(gamma[c] * (x[idx] - mu[c]) * rinv[c] + beta[c]));
}

// bn2 + elu fused with mean-pool accumulation
__global__ void bn_elu_pool(const float* __restrict__ x, const float* __restrict__ mu,
                            const float* __restrict__ rinv, const float* __restrict__ gamma,
                            const float* __restrict__ beta, const int* __restrict__ batch,
                            float* __restrict__ pools, int n, int cdim) {
  int idx = blockIdx.x * blockDim.x + threadIdx.x;
  if (idx >= n * cdim) return;
  int i = idx / cdim, c = idx - i * cdim;
  float v = eluf(gamma[c] * (x[idx] - mu[c]) * rinv[c] + beta[c]);
  atomicAdd(pools + (size_t)batch[i] * cdim + c, v);
}

__global__ void count_nodes(const int* __restrict__ batch, float* __restrict__ cnts, int n) {
  int i = blockIdx.x * blockDim.x + threadIdx.x;
  if (i < n) atomicAdd(cnts + batch[i], 1.f);
}

// ------------------------------- MLP head ----------------------------------

__global__ __launch_bounds__(64) void head_mlp(const float* __restrict__ pools,
                                               const float* __restrict__ cnts,
                                               const float* __restrict__ lw1,
                                               const float* __restrict__ lb1,
                                               const float* __restrict__ lw2,
                                               const float* __restrict__ lb2,
                                               float* __restrict__ out) {
  int g = blockIdx.x, c = threadIdx.x;
  __shared__ float pooled[HID], t[HID];
  float cnt = fmaxf(cnts[g], 1.f);
  pooled[c] = pools[(size_t)g * HID + c] / cnt;
  __syncthreads();
  float s = lb1[c];
  for (int k = 0; k < HID; ++k) s += pooled[k] * lw1[k * HID + c];
  t[c] = eluf(s) * lw2[c];
  __syncthreads();
  for (int o = 32; o > 0; o >>= 1) { if (c < o) t[c] += t[c + o]; __syncthreads(); }
  if (c == 0) out[g] = t[0] + lb2[0];
}

// ---------------------------------------------------------------------------

extern "C" void kernel_launch(void* const* d_in, const int* in_sizes, int n_in,
                              void* d_out, int out_size, void* d_ws, size_t ws_size,
                              hipStream_t stream) {
  const float* x      = (const float*)d_in[0];
  const int*   ei     = (const int*)  d_in[1];
  const int*   batch  = (const int*)  d_in[2];
  const float* W1l    = (const float*)d_in[3];
  const float* b1l    = (const float*)d_in[4];
  const float* W1r    = (const float*)d_in[5];
  const float* b1r    = (const float*)d_in[6];
  const float* att1   = (const float*)d_in[7];
  // bias1 (d_in[8]) cancels under BN: bn(x+b) == bn(x)
  const float* gamma1 = (const float*)d_in[9];
  const float* beta1  = (const float*)d_in[10];
  const float* W2l    = (const float*)d_in[11];
  const float* b2l    = (const float*)d_in[12];
  const float* W2r    = (const float*)d_in[13];
  const float* b2r    = (const float*)d_in[14];
  const float* att2   = (const float*)d_in[15];
  // bias2 (d_in[16]) cancels under BN
  const float* gamma2 = (const float*)d_in[17];
  const float* beta2  = (const float*)d_in[18];
  const float* lw1    = (const float*)d_in[19];
  const float* lb1    = (const float*)d_in[20];
  const float* lw2    = (const float*)d_in[21];
  const float* lb2    = (const float*)d_in[22];
  float* out = (float*)d_out;

  // workspace partition (256B aligned)
  char* p = (char*)d_ws;
  auto alloc = [&](size_t bytes) -> void* {
    void* r = (void*)p;
    p += (bytes + 255) & ~(size_t)255;
    return r;
  };
  bf16*  xb   = (bf16*) alloc((size_t)NN * INDIM * 2);
  bf16*  w1lp = (bf16*) alloc((size_t)INDIM * D1 * 2);
  bf16*  w1rp = (bf16*) alloc((size_t)INDIM * D1 * 2);
  float* xl1  = (float*)alloc((size_t)NN * D1 * 4);
  float* xr1  = (float*)alloc((size_t)NN * D1 * 4);   // reused as out1 after logits1
  float* lg1  = (float*)alloc((size_t)ETOT * HEADS * 4);
  float* a1   = (float*)alloc((size_t)ETOT * HEADS * 4);
  float* m1   = (float*)alloc((size_t)NN * HEADS * 4);
  float* den1 = (float*)alloc((size_t)NN * HEADS * 4);
  float* mu1  = (float*)alloc(D1 * 4);
  float* inv1 = (float*)alloc(D1 * 4);
  bf16*  hb   = (bf16*) alloc((size_t)NN * D1 * 2);
  bf16*  w2lp = (bf16*) alloc((size_t)D1 * HID * 2);
  bf16*  w2rp = (bf16*) alloc((size_t)D1 * HID * 2);
  float* xl2  = (float*)alloc((size_t)NN * HID * 4);
  float* xr2  = (float*)alloc((size_t)NN * HID * 4);  // reused as out2 after logits2
  float* lg2  = (float*)alloc((size_t)ETOT * 4);
  float* a2   = (float*)alloc((size_t)ETOT * 4);
  float* m2   = (float*)alloc((size_t)NN * 4);
  float* den2 = (float*)alloc((size_t)NN * 4);
  float* mu2  = (float*)alloc(HID * 4);
  float* inv2 = (float*)alloc(HID * 4);
  float* pools= (float*)alloc((size_t)GG * HID * 4);
  float* cnts = (float*)alloc((size_t)GG * 4);
  float* out1 = xr1;
  float* out2 = xr2;

  auto nb = [](long n, int b) { return (unsigned)((n + b - 1) / b); };
  const float NEGINF = -INFINITY;

  // ---- conv1 linear transforms (WMMA) ----
  cvt_f32_bf16<<<nb((long)NN * INDIM, 256), 256, 0, stream>>>(x, xb, NN * INDIM);
  pack_b_frag<<<nb((long)INDIM * D1, 256), 256, 0, stream>>>(W1l, w1lp, INDIM, D1);
  pack_b_frag<<<nb((long)INDIM * D1, 256), 256, 0, stream>>>(W1r, w1rp, INDIM, D1);
  {
    long waves = (long)(NN / 16) * (D1 / 64);                 // 12500 waves (4 n-tiles each)
    wmma_gemm_bias4<<<nb(waves, 4), 128, 0, stream>>>(xb, w1lp, b1l, xl1, NN, INDIM, D1);
    wmma_gemm_bias4<<<nb(waves, 4), 128, 0, stream>>>(xb, w1rp, b1r, xr1, NN, INDIM, D1);
  }

  // ---- conv1 attention / softmax / aggregate ----
  fill_f32<<<nb((long)NN * HEADS, 256), 256, 0, stream>>>(m1, NEGINF, NN * HEADS);
  edge_logits<<<nb((long)ETOT * HEADS, 256), 256, 0, stream>>>(xl1, xr1, att1, ei, lg1, m1, HEADS);
  fill_f32<<<nb((long)NN * HEADS, 256), 256, 0, stream>>>(den1, 0.f, NN * HEADS);
  edge_exp<<<nb((long)ETOT * HEADS, 256), 256, 0, stream>>>(lg1, m1, ei, a1, den1, HEADS);
  fill_f32<<<nb((long)NN * D1, 256), 256, 0, stream>>>(out1, 0.f, NN * D1);   // xr1 dead now
  edge_scatter<<<nb((long)ETOT * (D1 / 4), 256), 256, 0, stream>>>(a1, den1, xl1, ei, out1, HEADS);

  // ---- bn1 + elu -> bf16 ----
  bn_stats<<<D1, 256, 0, stream>>>(out1, NN, D1, mu1, inv1);
  bn_elu_bf16<<<nb((long)NN * D1, 256), 256, 0, stream>>>(out1, mu1, inv1, gamma1, beta1, hb, NN, D1);

  // ---- conv2 linear transforms (WMMA) ----
  pack_b_frag<<<nb((long)D1 * HID, 256), 256, 0, stream>>>(W2l, w2lp, D1, HID);
  pack_b_frag<<<nb((long)D1 * HID, 256), 256, 0, stream>>>(W2r, w2rp, D1, HID);
  {
    long waves = (long)(NN / 16) * (HID / 64);                // 3125 waves
    wmma_gemm_bias4<<<nb(waves, 4), 128, 0, stream>>>(hb, w2lp, b2l, xl2, NN, D1, HID);
    wmma_gemm_bias4<<<nb(waves, 4), 128, 0, stream>>>(hb, w2rp, b2r, xr2, NN, D1, HID);
  }

  // ---- conv2 attention / softmax / aggregate (heads=1) ----
  fill_f32<<<nb(NN, 256), 256, 0, stream>>>(m2, NEGINF, NN);
  edge_logits<<<nb(ETOT, 256), 256, 0, stream>>>(xl2, xr2, att2, ei, lg2, m2, 1);
  fill_f32<<<nb(NN, 256), 256, 0, stream>>>(den2, 0.f, NN);
  edge_exp<<<nb(ETOT, 256), 256, 0, stream>>>(lg2, m2, ei, a2, den2, 1);
  fill_f32<<<nb((long)NN * HID, 256), 256, 0, stream>>>(out2, 0.f, NN * HID); // xr2 dead now
  edge_scatter<<<nb((long)ETOT * (HID / 4), 256), 256, 0, stream>>>(a2, den2, xl2, ei, out2, 1);

  // ---- bn2 + elu fused with mean pool ----
  bn_stats<<<HID, 256, 0, stream>>>(out2, NN, HID, mu2, inv2);
  fill_f32<<<nb((long)GG * HID, 256), 256, 0, stream>>>(pools, 0.f, GG * HID);
  fill_f32<<<1, 64, 0, stream>>>(cnts, 0.f, GG);
  count_nodes<<<nb(NN, 256), 256, 0, stream>>>(batch, cnts, NN);
  bn_elu_pool<<<nb((long)NN * HID, 256), 256, 0, stream>>>(out2, mu2, inv2, gamma2, beta2, batch, pools, NN, HID);

  // ---- MLP head ----
  head_mlp<<<GG, 64, 0, stream>>>(pools, cnts, lw1, lb1, lw2, lb2, out);
}
